// ConfusionModule_60790967107721
// MI455X (gfx1250) — matrix operand order repaired and checked by reading, not compiled, not run on hardware
//
#include <hip/hip_runtime.h>

#define BQ 32
#define NB 128          // 4*B
#define L 512
#define E 512
#define SCALE_F 0.04419417382f   // 1/sqrt(512)
#define NEG_INF (-1e9f)

typedef __attribute__((ext_vector_type(16))) __bf16 v16bf;
typedef __attribute__((ext_vector_type(8)))  __bf16 v8bf;
typedef __attribute__((ext_vector_type(8)))  float  v8f;

#define ASTRIDE 40   // 64-row A tile, 32 K + pad (80B rows, 16B aligned)
#define BSTRIDE 40   // 128-row B^T tile, 32 K + pad

// ---- WMMA fragment loads from LDS ------------------------------------------
// A (16x32 bf16): lanes 0-15 -> row=lane, K 0..7 & 16..23 ; lanes 16-31 -> K 8..15 & 24..31
__device__ __forceinline__ v16bf load_a_frag(const __bf16* s, int lane, int mwave) {
  const int row = lane & 15;
  const int kb  = (lane >> 4) << 3;
  const __bf16* p = s + (mwave + row) * ASTRIDE + kb;
  v8bf lo = *(const v8bf*)p;
  v8bf hi = *(const v8bf*)(p + 16);
  v16bf a;
#pragma unroll
  for (int q = 0; q < 8; ++q) { a[q] = lo[q]; a[8 + q] = hi[q]; }
  return a;
}

// B (32x16 bf16): lane n<16 -> col=n, K 0..7 & 16..23 ; lane n>=16 -> col=n-16, K 8..15 & 24..31
// LDS holds B^T row-major: Bt[n][k]
__device__ __forceinline__ v16bf load_b_frag(const __bf16* s, int lane, int ntile) {
  const int col = lane & 15;
  const int kh  = (lane >> 4) << 3;
  const __bf16* p = s + (ntile * 16 + col) * BSTRIDE + kh;
  v8bf lo = *(const v8bf*)p;
  v8bf hi = *(const v8bf*)(p + 16);
  v16bf b;
#pragma unroll
  for (int q = 0; q < 8; ++q) { b[q] = lo[q]; b[8 + q] = hi[q]; }
  return b;
}

// split-bf16 compensated product: (ah+al)*(bh+bl) ~ ah*bh + ah*bl + al*bh
__device__ __forceinline__ v8f wmma3(v16bf ah, v16bf al, v16bf bh, v16bf bl, v8f c) {
  c = __builtin_amdgcn_wmma_f32_16x16x32_bf16(false, ah, false, bh, (short)0, c, false, false);
  c = __builtin_amdgcn_wmma_f32_16x16x32_bf16(false, ah, false, bl, (short)0, c, false, false);
  c = __builtin_amdgcn_wmma_f32_16x16x32_bf16(false, al, false, bh, (short)0, c, false, false);
  return c;
}

__device__ __forceinline__ void bsplit(float v, __bf16* ph, __bf16* pl) {
  __bf16 h = (__bf16)v;
  *ph = h;
  *pl = (__bf16)(v - (float)h);
}

// ---- K1: S = (x1[i%32] @ x2[i]^T) * scale, masked --------------------------
__global__ __launch_bounds__(256) void scores_kernel(
    const float* __restrict__ x1, const float* __restrict__ x2,
    const long long* __restrict__ x1_len, const long long* __restrict__ x2_len,
    float* __restrict__ attn) {
  __shared__ __bf16 Ah[2][64 * ASTRIDE], Al[2][64 * ASTRIDE];
  __shared__ __bf16 Bh[2][128 * BSTRIDE], Bl[2][128 * BSTRIDE];

  const int i  = blockIdx.x >> 5;        // batch 0..127
  const int t  = blockIdx.x & 31;
  const int m0 = (t >> 2) << 6;          // 64-row tile
  const int n0 = (t & 3) << 7;           // 128-col tile

  const float* Aq = x1 + (size_t)(i & (BQ - 1)) * L * E;
  const float* Kv = x2 + (size_t)i * L * E;

  const int tid  = threadIdx.x;
  const int wave = tid >> 5;
  const int lane = tid & 31;
  const int mw = (wave >> 1) << 4;       // {0,16,32,48}
  const int nw = (wave & 1) << 6;        // {0,64}

  const int srow = tid >> 2;             // 0..63
  const int scol = (tid & 3) << 3;       // 0,8,16,24

  auto stage = [&](int buf, int k0) {
    { // stage A (64x32)
      const float* g = Aq + (size_t)(m0 + srow) * E + k0 + scol;
      if (k0 + 32 < E) __builtin_prefetch(g + 32, 0, 0);
      __bf16* ph = &Ah[buf][srow * ASTRIDE + scol];
      __bf16* pl = &Al[buf][srow * ASTRIDE + scol];
#pragma unroll
      for (int q = 0; q < 8; ++q) bsplit(g[q], ph + q, pl + q);
    }
#pragma unroll
    for (int rep = 0; rep < 2; ++rep) {  // stage B^T (128 x 32) = x2 rows directly
      const int n = srow + (rep << 6);
      const float* g = Kv + (size_t)(n0 + n) * E + k0 + scol;
      if (k0 + 32 < E) __builtin_prefetch(g + 32, 0, 0);
      __bf16* ph = &Bh[buf][n * BSTRIDE + scol];
      __bf16* pl = &Bl[buf][n * BSTRIDE + scol];
#pragma unroll
      for (int q = 0; q < 8; ++q) bsplit(g[q], ph + q, pl + q);
    }
  };

  v8f acc[4];
#pragma unroll
  for (int tt = 0; tt < 4; ++tt) acc[tt] = (v8f)(0.0f);

  stage(0, 0);
  __syncthreads();
  for (int k0 = 0, it = 0; k0 < E; k0 += 32, ++it) {
    const int cur = it & 1;
    if (k0 + 32 < E) stage(cur ^ 1, k0 + 32);
    v16bf ah = load_a_frag(&Ah[cur][0], lane, mw);
    v16bf al = load_a_frag(&Al[cur][0], lane, mw);
#pragma unroll
    for (int tt = 0; tt < 4; ++tt) {
      v16bf bh = load_b_frag(&Bh[cur][0], lane, (nw >> 4) + tt);
      v16bf bl = load_b_frag(&Bl[cur][0], lane, (nw >> 4) + tt);
      acc[tt] = wmma3(ah, al, bh, bl, acc[tt]);
    }
    __syncthreads();
  }

  const long long qlen = x1_len[i & (BQ - 1)];
  const long long klen = x2_len[i];
  float* S = attn + (size_t)i * L * L;
  const int half = lane >> 4;
  const int nin  = lane & 15;
#pragma unroll
  for (int tt = 0; tt < 4; ++tt) {
    const int n = n0 + nw + tt * 16 + nin;
    const bool kv = (long long)n < klen;
#pragma unroll
    for (int r = 0; r < 8; ++r) {
      const int m = m0 + mw + (half << 3) + r;
      const float v = (kv && (long long)m < qlen) ? acc[tt][r] * SCALE_F : NEG_INF;
      S[(size_t)m * L + n] = v;
    }
  }
}

// ---- K2: row softmax (one wave per 512-wide row) ---------------------------
__global__ __launch_bounds__(256) void softmax_kernel(float* __restrict__ attn) {
  const int wave = threadIdx.x >> 5;
  const int lane = threadIdx.x & 31;
  const int row  = blockIdx.x * 8 + wave;
  float* R = attn + (size_t)row * L;
  float v[16];
#pragma unroll
  for (int q = 0; q < 16; ++q) v[q] = R[lane + (q << 5)];
  float m = v[0];
#pragma unroll
  for (int q = 1; q < 16; ++q) m = fmaxf(m, v[q]);
#pragma unroll
  for (int off = 16; off > 0; off >>= 1) m = fmaxf(m, __shfl_xor(m, off, 32));
  float s = 0.f;
#pragma unroll
  for (int q = 0; q < 16; ++q) { v[q] = __expf(v[q] - m); s += v[q]; }
#pragma unroll
  for (int off = 16; off > 0; off >>= 1) s += __shfl_xor(s, off, 32);
  const float inv = 1.0f / s;
#pragma unroll
  for (int q = 0; q < 16; ++q) R[lane + (q << 5)] = v[q] * inv;
}

// ---- K3: x1_att = attn @ x2 ------------------------------------------------
__global__ __launch_bounds__(256) void attnv_kernel(
    const float* __restrict__ attn, const float* __restrict__ x2,
    float* __restrict__ att_out) {
  __shared__ __bf16 Ah[2][64 * ASTRIDE], Al[2][64 * ASTRIDE];
  __shared__ __bf16 Bh[2][128 * BSTRIDE], Bl[2][128 * BSTRIDE];

  const int i  = blockIdx.x >> 5;
  const int t  = blockIdx.x & 31;
  const int m0 = (t >> 2) << 6;
  const int n0 = (t & 3) << 7;

  const float* Aq = attn + (size_t)i * L * L;
  const float* Kv = x2 + (size_t)i * L * E;

  const int tid  = threadIdx.x;
  const int wave = tid >> 5;
  const int lane = tid & 31;
  const int mw = (wave >> 1) << 4;
  const int nw = (wave & 1) << 6;

  const int srow = tid >> 2;
  const int scol = (tid & 3) << 3;

  auto stage = [&](int buf, int k0) {
    { // stage A from attn
      const float* g = Aq + (size_t)(m0 + srow) * L + k0 + scol;
      if (k0 + 32 < L) __builtin_prefetch(g + 32, 0, 0);
      __bf16* ph = &Ah[buf][srow * ASTRIDE + scol];
      __bf16* pl = &Al[buf][srow * ASTRIDE + scol];
#pragma unroll
      for (int q = 0; q < 8; ++q) bsplit(g[q], ph + q, pl + q);
    }
    // stage B^T[n=e][k=l2] = x2[l2][e] : transpose during staging
    for (int idx = tid; idx < 32 * 128; idx += 256) {
      const int kr = idx >> 7, n = idx & 127;
      const float* g = &Kv[(size_t)(k0 + kr) * E + n0 + n];
      if (k0 + 32 < L) __builtin_prefetch(g + 32 * E, 0, 0);
      bsplit(*g, &Bh[buf][n * BSTRIDE + kr], &Bl[buf][n * BSTRIDE + kr]);
    }
  };

  v8f acc[4];
#pragma unroll
  for (int tt = 0; tt < 4; ++tt) acc[tt] = (v8f)(0.0f);

  stage(0, 0);
  __syncthreads();
  for (int k0 = 0, it = 0; k0 < L; k0 += 32, ++it) {
    const int cur = it & 1;
    if (k0 + 32 < L) stage(cur ^ 1, k0 + 32);
    v16bf ah = load_a_frag(&Ah[cur][0], lane, mw);
    v16bf al = load_a_frag(&Al[cur][0], lane, mw);
#pragma unroll
    for (int tt = 0; tt < 4; ++tt) {
      v16bf bh = load_b_frag(&Bh[cur][0], lane, (nw >> 4) + tt);
      v16bf bl = load_b_frag(&Bl[cur][0], lane, (nw >> 4) + tt);
      acc[tt] = wmma3(ah, al, bh, bl, acc[tt]);
    }
    __syncthreads();
  }

  float* O = att_out + (size_t)i * L * E;
  const int half = lane >> 4;
  const int nin  = lane & 15;
#pragma unroll
  for (int tt = 0; tt < 4; ++tt) {
    const int n = n0 + nw + tt * 16 + nin;
#pragma unroll
    for (int r = 0; r < 8; ++r) {
      const int m = m0 + mw + (half << 3) + r;
      O[(size_t)m * E + n] = acc[tt][r];
    }
  }
}

// ---- K4: fus = relu(cat @ W + b), fused mean/max pooling -------------------
__global__ __launch_bounds__(256) void fusion_kernel(
    const float* __restrict__ x1, const float* __restrict__ att,
    const float* __restrict__ W, const float* __restrict__ bias,
    float* __restrict__ pool_sum, float* __restrict__ pool_max) {
  __shared__ __bf16 Ah[2][64 * ASTRIDE], Al[2][64 * ASTRIDE];
  __shared__ __bf16 Bh[2][128 * BSTRIDE], Bl[2][128 * BSTRIDE];

  const int i  = blockIdx.x >> 5;
  const int t  = blockIdx.x & 31;
  const int m0 = (t >> 2) << 6;
  const int j0 = (t & 3) << 7;

  const float* Xq = x1 + (size_t)(i & (BQ - 1)) * L * E;
  const float* Av = att + (size_t)i * L * E;

  const int tid  = threadIdx.x;
  const int wave = tid >> 5;
  const int lane = tid & 31;
  const int mw = (wave >> 1) << 4;
  const int nw = (wave & 1) << 6;

  const int srow = tid >> 2;
  const int scol = (tid & 3) << 3;

  auto stage = [&](int buf, int k0) {
    const int  e0  = k0 & (E - 1);
    const bool mul = (k0 >= E);
    { // stage A: cat = (k<512) ? x1-att : x1*att, built on the fly
      const float* gx = Xq + (size_t)(m0 + srow) * E + e0 + scol;
      const float* ga = Av + (size_t)(m0 + srow) * E + e0 + scol;
      __bf16* ph = &Ah[buf][srow * ASTRIDE + scol];
      __bf16* pl = &Al[buf][srow * ASTRIDE + scol];
#pragma unroll
      for (int q = 0; q < 8; ++q) {
        const float xv = gx[q], av = ga[q];
        bsplit(mul ? (xv * av) : (xv - av), ph + q, pl + q);
      }
    }
    // stage B^T[n=j][k=e] = W[e][j] : transpose during staging
    for (int idx = tid; idx < 32 * 128; idx += 256) {
      const int kr = idx >> 7, n = idx & 127;
      const float* g = &W[(size_t)(k0 + kr) * E + j0 + n];
      if (k0 + 32 < 2 * E) __builtin_prefetch(g + 32 * E, 0, 0);
      bsplit(*g, &Bh[buf][n * BSTRIDE + kr], &Bl[buf][n * BSTRIDE + kr]);
    }
  };

  v8f acc[4];
#pragma unroll
  for (int tt = 0; tt < 4; ++tt) acc[tt] = (v8f)(0.0f);

  stage(0, 0);
  __syncthreads();
  for (int k0 = 0, it = 0; k0 < 2 * E; k0 += 32, ++it) {
    const int cur = it & 1;
    if (k0 + 32 < 2 * E) stage(cur ^ 1, k0 + 32);
    v16bf ah = load_a_frag(&Ah[cur][0], lane, mw);
    v16bf al = load_a_frag(&Al[cur][0], lane, mw);
#pragma unroll
    for (int tt = 0; tt < 4; ++tt) {
      v16bf bh = load_b_frag(&Bh[cur][0], lane, (nw >> 4) + tt);
      v16bf bl = load_b_frag(&Bl[cur][0], lane, (nw >> 4) + tt);
      acc[tt] = wmma3(ah, al, bh, bl, acc[tt]);
    }
    __syncthreads();
  }

  // epilogue: bias + relu + partial mean/max pooling over the 16 rows of this strip
  const int half = lane >> 4;
  const int nin  = lane & 15;
#pragma unroll
  for (int tt = 0; tt < 4; ++tt) {
    const int j = j0 + nw + tt * 16 + nin;
    const float bb = bias[j];
    float s = 0.f, mx = 0.f;
#pragma unroll
    for (int r = 0; r < 8; ++r) {
      const float v = fmaxf(acc[tt][r] + bb, 0.f);
      s += v;
      mx = fmaxf(mx, v);
    }
    s += __shfl_xor(s, 16, 32);                 // combine rows 0-7 with 8-15
    mx = fmaxf(mx, __shfl_xor(mx, 16, 32));
    if (half == 0) {
      atomicAdd(pool_sum + (size_t)i * E + j, s);
      // relu output >= 0, so uint ordering == float ordering
      atomicMax((unsigned int*)(pool_max + (size_t)i * E + j), __float_as_uint(mx));
    }
  }
}

// ---- K5: out = relu(pooled @ out_w + out_b) --------------------------------
__global__ __launch_bounds__(256) void out_kernel(
    const float* __restrict__ pool_sum, const float* __restrict__ pool_max,
    const float* __restrict__ W2, const float* __restrict__ b2,
    float* __restrict__ out) {
  __shared__ float P[8 * E];     // 4096-wide pooled row for this batch
  const int b = blockIdx.x;
  const int tid = threadIdx.x;
  for (int idx = tid; idx < 8 * E; idx += 256) {
    const int hf = idx >> 11;    // 0 = mean part, 1 = max part
    const int k  = idx & 2047;
    const int d  = k >> 9, e = k & 511;
    const int ii = (b << 2) + d;
    P[idx] = hf ? pool_max[(size_t)ii * E + e]
                : pool_sum[(size_t)ii * E + e] * (1.0f / (float)L);
  }
  __syncthreads();
  float acc[4];
#pragma unroll
  for (int q = 0; q < 4; ++q) acc[q] = b2[tid + (q << 8)];
  for (int k = 0; k < 8 * E; ++k) {
    const float p = P[k];
    const float* w = W2 + (size_t)k * (2 * E);
#pragma unroll
    for (int q = 0; q < 4; ++q) acc[q] += p * w[tid + (q << 8)];
  }
#pragma unroll
  for (int q = 0; q < 4; ++q)
    out[(size_t)b * (2 * E) + tid + (q << 8)] = fmaxf(acc[q], 0.f);
}

extern "C" void kernel_launch(void* const* d_in, const int* in_sizes, int n_in,
                              void* d_out, int out_size, void* d_ws, size_t ws_size,
                              hipStream_t stream) {
  const float*     x1       = (const float*)d_in[0];
  const float*     x2       = (const float*)d_in[1];
  const long long* x1_len   = (const long long*)d_in[2];
  const long long* x2_len   = (const long long*)d_in[3];
  const float*     fusion_w = (const float*)d_in[4];
  const float*     fusion_b = (const float*)d_in[5];
  const float*     out_w    = (const float*)d_in[6];
  const float*     out_b    = (const float*)d_in[7];

  float* out  = (float*)d_out;                       // [32, 1024]
  float* attn = out + (size_t)BQ * 2 * E;            // [128, 512, 512]

  float* att_ws   = (float*)d_ws;                    // [128, 512, 512] f32
  float* pool_sum = att_ws + (size_t)NB * L * E;     // [128, 512]
  float* pool_max = pool_sum + (size_t)NB * E;       // [128, 512]

  hipMemsetAsync(pool_sum, 0, (size_t)2 * NB * E * sizeof(float), stream);

  const dim3 blk(256);
  scores_kernel <<<dim3(NB * 32),    blk, 0, stream>>>(x1, x2, x1_len, x2_len, attn);
  softmax_kernel<<<dim3(NB * L / 8), blk, 0, stream>>>(attn);
  attnv_kernel  <<<dim3(NB * 32),    blk, 0, stream>>>(attn, x2, att_ws);
  fusion_kernel <<<dim3(NB * 32),    blk, 0, stream>>>(x1, att_ws, fusion_w, fusion_b,
                                                       pool_sum, pool_max);
  out_kernel    <<<dim3(BQ),         blk, 0, stream>>>(pool_sum, pool_max, out_w, out_b, out);
}
